// GRUNet_65798898975058
// MI455X (gfx1250) — matrix operand order repaired and checked by reading, not compiled, not run on hardware
//
#include <hip/hip_runtime.h>

// ---------------------------------------------------------------------------
// Problem constants (from reference)
// ---------------------------------------------------------------------------
#define NROWS 8192
#define DIM   512            // D
#define SDIM  1024           // S = 2D
#define HDIM  512            // H
#define H3    1536           // 3H
#define MAXPAD 3
#define EPSBN 1e-5f

// # of 16-column W_hh tiles pinned in LDS for the scan (of 96 total).
// 18 tiles * 16 cols * 512 K * 2B = 288 KB; + ~11 KB state = ~300 KB,
// within CDNA5's 320 KB per-workgroup LDS (ISA 00_overview §3.1).
#define CT_CACHED 18

typedef __bf16 bf16x8  __attribute__((ext_vector_type(8)));
typedef __bf16 v16bf   __attribute__((ext_vector_type(16)));
typedef float  v8f     __attribute__((ext_vector_type(8)));

// ---------------------------------------------------------------------------
// f32 -> bf16 convert (grid-stride)
// ---------------------------------------------------------------------------
__global__ void k_f32_to_bf16(const float* __restrict__ src,
                              __bf16* __restrict__ dst, int n) {
    for (int i = blockIdx.x * blockDim.x + threadIdx.x; i < n;
         i += gridDim.x * blockDim.x)
        dst[i] = (__bf16)src[i];
}

// ---------------------------------------------------------------------------
// WMMA bf16 GEMM:  C[M x Nout] = A[M x K] * W[Nout x K]^T  (+bias, opt tanh)
// One wave computes a 16x64 output strip (4 accumulators, A reused 4x).
// Fragment layouts used here (wave32; contiguous k, same for A and B):
//   A lane l (row = m0 + (l&15)):  elems 0..15 = K[k0 + hi*16 .. +15]
//   B lane l (col = n0 + (l&15)):  elems 0..15 = K[k0 + hi*16 .. +15]
//   C/D lane l, vgpr v: row = m0 + v + hi*8, col = n0 + (l&15)
//   (hi = l>>4)
// ---------------------------------------------------------------------------
__global__ __launch_bounds__(256)
void k_gemm_bf16(const __bf16* __restrict__ A, const __bf16* __restrict__ W,
                 const float* __restrict__ bias, float* __restrict__ C,
                 int M, int K, int Nout, int mode /*0=bias, 1=bias+tanh*/) {
    const int lane = threadIdx.x & 31;
    const int wave = blockIdx.x * (blockDim.x >> 5) + (threadIdx.x >> 5);
    const int tiles_m = M >> 4;
    const int tiles_n4 = Nout >> 6;
    if (wave >= tiles_m * tiles_n4) return;

    const int mt = wave % tiles_m;
    const int nt = wave / tiles_m;
    const int m0 = mt << 4;
    const int n0 = nt << 6;
    const int lo16 = lane & 15;
    const int hi = lane >> 4;

    const __bf16* Arow = A + (size_t)(m0 + lo16) * K;
    const __bf16* Wr0 = W + (size_t)(n0 + lo16) * K;
    const __bf16* Wr1 = Wr0 + (size_t)16 * K;
    const __bf16* Wr2 = Wr0 + (size_t)32 * K;
    const __bf16* Wr3 = Wr0 + (size_t)48 * K;

    v8f acc0 = {}, acc1 = {}, acc2 = {}, acc3 = {};

    for (int k0 = 0; k0 < K; k0 += 32) {
        v16bf a = *(const v16bf*)(Arow + k0 + hi * 16);
        const int boff = k0 + hi * 16;
        v16bf b0 = *(const v16bf*)(Wr0 + boff);
        v16bf b1 = *(const v16bf*)(Wr1 + boff);
        v16bf b2 = *(const v16bf*)(Wr2 + boff);
        v16bf b3 = *(const v16bf*)(Wr3 + boff);

        acc0 = __builtin_amdgcn_wmma_f32_16x16x32_bf16(false, a, false, b0,
                                                       (short)0, acc0, false, false);
        acc1 = __builtin_amdgcn_wmma_f32_16x16x32_bf16(false, a, false, b1,
                                                       (short)0, acc1, false, false);
        acc2 = __builtin_amdgcn_wmma_f32_16x16x32_bf16(false, a, false, b2,
                                                       (short)0, acc2, false, false);
        acc3 = __builtin_amdgcn_wmma_f32_16x16x32_bf16(false, a, false, b3,
                                                       (short)0, acc3, false, false);
    }

    const int rbase = m0 + hi * 8;
    v8f accs[4] = {acc0, acc1, acc2, acc3};
#pragma unroll
    for (int j = 0; j < 4; ++j) {
        const int col = n0 + j * 16 + lo16;
        const float bb = bias[col];
#pragma unroll
        for (int v = 0; v < 8; ++v) {
            float val = accs[j][v] + bb;
            if (mode == 1) val = tanhf(val);
            C[(size_t)(rbase + v) * Nout + col] = val;
        }
    }
}

// ---------------------------------------------------------------------------
// Per-column batch statistics (training-mode BN, biased variance)
// ---------------------------------------------------------------------------
__global__ __launch_bounds__(256)
void k_colstats(const float* __restrict__ X, int M, int Ccols,
                float* __restrict__ mean, float* __restrict__ rstd) {
    const int c = blockIdx.x;
    float s = 0.f, sq = 0.f;
    for (int r = threadIdx.x; r < M; r += blockDim.x) {
        float v = X[(size_t)r * Ccols + c];
        s += v;
        sq += v * v;
    }
    __shared__ float ss[256];
    __shared__ float ssq[256];
    ss[threadIdx.x] = s;
    ssq[threadIdx.x] = sq;
    __syncthreads();
    for (int o = 128; o > 0; o >>= 1) {
        if (threadIdx.x < o) {
            ss[threadIdx.x] += ss[threadIdx.x + o];
            ssq[threadIdx.x] += ssq[threadIdx.x + o];
        }
        __syncthreads();
    }
    if (threadIdx.x == 0) {
        float mu = ss[0] / (float)M;
        float var = ssq[0] / (float)M - mu * mu;
        mean[c] = mu;
        rstd[c] = rsqrtf(var + EPSBN);
    }
}

// ---------------------------------------------------------------------------
// Apply BN (+ optional ReLU) and emit bf16 for the next WMMA GEMM
// ---------------------------------------------------------------------------
__global__ void k_bn_act_bf16(const float* __restrict__ X,
                              const float* __restrict__ mean,
                              const float* __restrict__ rstd,
                              const float* __restrict__ g,
                              const float* __restrict__ be,
                              __bf16* __restrict__ out, int n, int Ccols,
                              int relu) {
    for (int i = blockIdx.x * blockDim.x + threadIdx.x; i < n;
         i += gridDim.x * blockDim.x) {
        int c = i % Ccols;
        float v = (X[i] - mean[c]) * rstd[c] * g[c] + be[c];
        if (relu) v = fmaxf(v, 0.f);
        out[i] = (__bf16)v;
    }
}

// ---------------------------------------------------------------------------
// Sequential GRU scan, WMMA edition.
//
// One 1024-thread workgroup (32 wave32s, 8 waves/SIMD for latency hiding).
// Per GRU cell, gh = W_hh @ h + b_hh is computed on the matrix pipe:
//   * h (bf16, in LDS) is row 0 of the A fragment (rows 1..15 zero);
//   * W_hh rows (bf16, row-major) are B fragments — same addressing as the
//     GEMM's weight operand, so no transpose is needed;
//   * each wave owns 3 of the 96 16-column output tiles (16 K-steps each);
//   * D row 0 (vgpr0, lanes hi==0) is scattered to gh[] in LDS.
// The first CT_CACHED tiles of W_hh are pinned in LDS (CDNA5: 320 KB/WG),
// the rest stream from the 192 MB L2 (W_hh bf16 = 1.5 MB, fully resident).
// Gate math and h/s/c state remain f32; thread i owns hidden unit i.
// ---------------------------------------------------------------------------
__device__ __forceinline__ float sigmf(float x) {
    return 1.f / (1.f + __expf(-x));
}

__global__ __launch_bounds__(1024)
void k_scan(const float* __restrict__ xg_all, const int* __restrict__ zero_pad,
            const int* __restrict__ broad_id, const __bf16* __restrict__ Whh_b,
            const float* __restrict__ b_ih, const float* __restrict__ b_hh,
            float* __restrict__ outs) {
    extern __shared__ __align__(16) char smem[];
    float*  h32  = (float*)smem;                       // 512 f32
    float*  sacc = h32 + HDIM;                         // 512 f32
    float*  gh   = sacc + HDIM;                        // 1536 f32
    __bf16* hb   = (__bf16*)(gh + H3);                 // 512 bf16
    __bf16* ldsW = hb + HDIM;                          // CT_CACHED*16*512 bf16

    const int tid  = threadIdx.x;
    const int wv   = tid >> 5;        // wave 0..31
    const int lane = tid & 31;
    const int lo16 = lane & 15;
    const int hi   = lane >> 4;

    // --- stage 1: pin first CT_CACHED column tiles of W_hh in LDS ---
    {
        const int nwords = CT_CACHED * 16 * HDIM / 2;  // bf16 pairs
        const unsigned int* src = (const unsigned int*)Whh_b;
        unsigned int* dst = (unsigned int*)ldsW;
        for (int t = tid; t < nwords; t += 1024) dst[t] = src[t];
    }

    float bhr = 0.f, bhz = 0.f, bhn = 0.f;
    float bir = 0.f, biz = 0.f, bin_ = 0.f;
    float cnt = 0.f;
    if (tid < HDIM) {
        bhr = b_hh[tid]; bhz = b_hh[HDIM + tid]; bhn = b_hh[2 * HDIM + tid];
        bir = b_ih[tid]; biz = b_ih[HDIM + tid]; bin_ = b_ih[2 * HDIM + tid];
        h32[tid] = 0.f;
        hb[tid] = (__bf16)0.f;
        sacc[tid] = 0.f;
    }
    __syncthreads();

    // WMMA matvec phase: fill gh[0..1535] from current hb (all 1024 threads)
    auto matvec = [&]() {
#pragma unroll
        for (int j = 0; j < 3; ++j) {
            const int ct = wv * 3 + j;          // column tile 0..95
            const int n0 = ct << 4;
            const __bf16* Wrow =
                (ct < CT_CACHED)
                    ? (ldsW + (size_t)(n0 + lo16) * HDIM)
                    : (Whh_b + (size_t)(n0 + lo16) * HDIM);
            v8f acc = {};
            for (int k0 = 0; k0 < HDIM; k0 += 32) {
                v16bf a = {};
                if (lo16 == 0) {  // only m==0 lanes carry h
                    a = *(const v16bf*)(hb + k0 + hi * 16);
                }
                v16bf b = *(const v16bf*)(Wrow + k0 + hi * 16);
                acc = __builtin_amdgcn_wmma_f32_16x16x32_bf16(
                    false, a, false, b, (short)0, acc, false, false);
            }
            if (hi == 0)  // D row 0 lives in vgpr0 of lanes 0..15
                gh[n0 + lo16] = acc[0] + b_hh[n0 + lo16];
        }
    };

    for (int row = 0; row < NROWS; ++row) {
        const int zp = zero_pad[row];  // uniform across block

        // --- zero-pad GRU cells (input gates = b_ih) ---
        for (int k = 0; k < MAXPAD; ++k) {
            if (k < zp) {  // uniform branch
                matvec();
                __syncthreads();
                if (tid < HDIM) {
                    const float r = sigmf(bir + gh[tid]);
                    const float z = sigmf(biz + gh[HDIM + tid]);
                    const float n = tanhf(bin_ + r * gh[2 * HDIM + tid]);
                    const float hn = (1.f - z) * n + z * h32[tid];
                    h32[tid] = hn;
                    hb[tid] = (__bf16)hn;
                }
                __syncthreads();
            }
        }

        // --- real GRU cell ---
        matvec();
        __syncthreads();
        if (tid < HDIM) {
            const float* xg = xg_all + (size_t)row * H3;
            const float r = sigmf(xg[tid] + gh[tid]);
            const float z = sigmf(xg[HDIM + tid] + gh[HDIM + tid]);
            const float n = tanhf(xg[2 * HDIM + tid] + r * gh[2 * HDIM + tid]);
            const float htmp = (1.f - z) * n + z * h32[tid];
            outs[(size_t)row * HDIM + tid] = htmp;

            const float s2 = sacc[tid] + htmp;
            const float c2 = cnt + 1.f;
            const bool bnd =
                (row == NROWS - 1) || (broad_id[row] != broad_id[row + 1]);
            float hn;
            if (bnd) {
                hn = s2 / c2;
                sacc[tid] = 0.f;
                cnt = 0.f;
            } else {
                hn = htmp;
                sacc[tid] = s2;
                cnt = c2;
            }
            h32[tid] = hn;
            hb[tid] = (__bf16)hn;
            if (row + 1 < NROWS)  // warm L2/L0 for next row's input gates
                __builtin_prefetch(xg_all + (size_t)(row + 1) * H3 + tid, 0, 1);
        }
        __syncthreads();
    }
}

// ---------------------------------------------------------------------------
// Final head: out[r] = tanh( dot(y[r, :256], W3) + b3 )
// ---------------------------------------------------------------------------
__global__ void k_head(const float* __restrict__ y, const float* __restrict__ W3,
                       const float* __restrict__ b3, float* __restrict__ out,
                       int nrows) {
    const int r = blockIdx.x * blockDim.x + threadIdx.x;
    if (r >= nrows) return;
    const float* yr = y + (size_t)r * (HDIM / 2);
    float acc = 0.f;
#pragma unroll 8
    for (int k = 0; k < HDIM / 2; ++k) acc += yr[k] * W3[k];
    out[r] = tanhf(acc + b3[0]);
}

// ---------------------------------------------------------------------------
// Host launcher
// ---------------------------------------------------------------------------
extern "C" void kernel_launch(void* const* d_in, const int* in_sizes, int n_in,
                              void* d_out, int out_size, void* d_ws,
                              size_t ws_size, hipStream_t stream) {
    (void)in_sizes; (void)n_in; (void)out_size; (void)ws_size;

    const float* x       = (const float*)d_in[0];
    const int*   zero_pad= (const int*)  d_in[1];
    const int*   broad_id= (const int*)  d_in[2];
    const float* W1      = (const float*)d_in[3];
    const float* b1      = (const float*)d_in[4];
    const float* g1      = (const float*)d_in[5];
    const float* be1     = (const float*)d_in[6];
    const float* W_ih    = (const float*)d_in[7];
    const float* W_hh    = (const float*)d_in[8];
    const float* b_ih    = (const float*)d_in[9];
    const float* b_hh    = (const float*)d_in[10];
    const float* g2      = (const float*)d_in[11];
    const float* be2     = (const float*)d_in[12];
    const float* W2      = (const float*)d_in[13];
    const float* b2      = (const float*)d_in[14];
    const float* W3      = (const float*)d_in[15];
    const float* b3      = (const float*)d_in[16];
    float* out = (float*)d_out;

    // ---- workspace carve-up (256B aligned), with liveness-based reuse ----
    char* p = (char*)d_ws;
    auto take = [&](size_t bytes) -> char* {
        char* r = p;
        p += (bytes + 255) & ~(size_t)255;
        return r;
    };
    __bf16* x_b   = (__bf16*)take((size_t)NROWS * DIM * 2);      // 8 MB
    __bf16* W1b   = (__bf16*)take((size_t)SDIM * DIM * 2);       // 1 MB
    __bf16* Wihb  = (__bf16*)take((size_t)H3 * SDIM * 2);        // 3 MB
    __bf16* W2b   = (__bf16*)take((size_t)(HDIM / 2) * HDIM * 2);// .25 MB
    __bf16* Whhb  = (__bf16*)take((size_t)H3 * HDIM * 2);        // 1.5 MB
    // R1: xf_pre [N x S] f32 (32MB) then reused as xg_all [N x 3H] f32 (48MB)
    char*   R1    = take((size_t)NROWS * H3 * 4);                // 48 MB
    float*  xf_pre = (float*)R1;
    float*  xg_all = (float*)R1;
    // R2: xf_bf16 [N x S] (16MB) then reused as outs [N x H] f32 (16MB)
    char*   R2    = take((size_t)NROWS * SDIM * 2);              // 16 MB
    __bf16* xf_b  = (__bf16*)R2;
    float*  outs  = (float*)R2;
    __bf16* obn_b = (__bf16*)take((size_t)NROWS * HDIM * 2);     // 8 MB
    float*  ymid  = (float*)take((size_t)NROWS * (HDIM / 2) * 4);// 8 MB
    float*  mean1 = (float*)take(SDIM * 4);
    float*  rstd1 = (float*)take(SDIM * 4);
    float*  mean2 = (float*)take(HDIM * 4);
    float*  rstd2 = (float*)take(HDIM * 4);

    // ---- 0. precision conversion of static operands ----
    k_f32_to_bf16<<<2048, 256, 0, stream>>>(x, x_b, NROWS * DIM);
    k_f32_to_bf16<<<1024, 256, 0, stream>>>(W1, W1b, SDIM * DIM);
    k_f32_to_bf16<<<2048, 256, 0, stream>>>(W_ih, Wihb, H3 * SDIM);
    k_f32_to_bf16<<<512, 256, 0, stream>>>(W2, W2b, (HDIM / 2) * HDIM);
    k_f32_to_bf16<<<1024, 256, 0, stream>>>(W_hh, Whhb, H3 * HDIM);

    // ---- 1. xf_pre = x @ W1^T + b1  (WMMA) ----
    {
        int waves = (NROWS / 16) * (SDIM / 64);  // 8192
        k_gemm_bf16<<<waves / 8, 256, 0, stream>>>(x_b, W1b, b1, xf_pre,
                                                   NROWS, DIM, SDIM, 0);
    }
    // ---- 2. BN1 stats + apply + ReLU -> bf16 ----
    k_colstats<<<SDIM, 256, 0, stream>>>(xf_pre, NROWS, SDIM, mean1, rstd1);
    k_bn_act_bf16<<<2048, 256, 0, stream>>>(xf_pre, mean1, rstd1, g1, be1,
                                            xf_b, NROWS * SDIM, SDIM, 1);
    // ---- 3. xg_all = xf @ W_ih^T + b_ih  (WMMA; overwrites dead xf_pre) ----
    {
        int waves = (NROWS / 16) * (H3 / 64);  // 12288
        k_gemm_bf16<<<waves / 8, 256, 0, stream>>>(xf_b, Wihb, b_ih, xg_all,
                                                   NROWS, SDIM, H3, 0);
    }
    // ---- 4. sequential GRU scan, WMMA matvec + 300 KB LDS (one WGP) ----
    {
        const size_t smem = (size_t)HDIM * 4 * 2      // h32, sacc
                          + (size_t)H3 * 4            // gh
                          + (size_t)HDIM * 2          // hb
                          + (size_t)CT_CACHED * 16 * HDIM * 2;  // W_hh cache
        (void)hipFuncSetAttribute((const void*)k_scan,
                                  hipFuncAttributeMaxDynamicSharedMemorySize,
                                  (int)smem);
        k_scan<<<1, 1024, smem, stream>>>(xg_all, zero_pad, broad_id, Whhb,
                                          b_ih, b_hh, outs);
    }
    // ---- 5. BN2 stats + apply -> bf16 ----
    k_colstats<<<HDIM, 256, 0, stream>>>(outs, NROWS, HDIM, mean2, rstd2);
    k_bn_act_bf16<<<2048, 256, 0, stream>>>(outs, mean2, rstd2, g2, be2, obn_b,
                                            NROWS * HDIM, HDIM, 0);
    // ---- 6. y = tanh(obn @ W2^T + b2)  (WMMA with tanh epilogue) ----
    {
        int waves = (NROWS / 16) * ((HDIM / 2) / 64);  // 2048
        k_gemm_bf16<<<waves / 8, 256, 0, stream>>>(obn_b, W2b, b2, ymid,
                                                   NROWS, HDIM, HDIM / 2, 1);
    }
    // ---- 7. head: tanh(y @ W3^T + b3) ----
    k_head<<<NROWS / 256, 256, 0, stream>>>(ymid, W3, b3, out, NROWS);
}
